// CfDtAn_75814762709726
// MI455X (gfx1250) — matrix operand order repaired and checked by reading, not compiled, not run on hardware
//
#include <hip/hip_runtime.h>
#include <hip/hip_bf16.h>

typedef __attribute__((ext_vector_type(16))) _Float16 v16h;
typedef __attribute__((ext_vector_type(8)))  _Float16 v8h;
typedef __attribute__((ext_vector_type(8)))  float    v8f;

// ---------------------------------------------------------------------------
// Fragment loaders (wave32; this code's contiguous k order, identical for A and B)
// A 16x32 f16: lane row M = lane%16; a[i]: K = k0 + i + 16*hi
// B 32x16 f16: lane col N = lane%16; b[i]: K = k0 + i + 16*hi
// ---------------------------------------------------------------------------
__device__ __forceinline__ v16h load_a16(const _Float16* __restrict__ row, int k0, int hi) {
    v8h c0 = *(const v8h*)(row + k0 + hi * 16);
    v8h c1 = *(const v8h*)(row + k0 + hi * 16 + 8);
    v16h a;
#pragma unroll
    for (int i = 0; i < 8; ++i) { a[i] = c0[i]; a[8 + i] = c1[i]; }
    return a;
}
__device__ __forceinline__ v16h load_b16_contig(const _Float16* __restrict__ row, int k0, int hi) {
    v8h c0 = *(const v8h*)(row + k0 + hi * 16);
    v8h c1 = *(const v8h*)(row + k0 + hi * 16 + 8);
    v16h b;
#pragma unroll
    for (int i = 0; i < 8; ++i) { b[i] = c0[i]; b[8 + i] = c1[i]; }
    return b;
}

// ---------------------------------------------------------------------------
// Prep: fold conv bias + BN into per-channel scale/shift
// ---------------------------------------------------------------------------
__global__ void prep_scales(
    const float* c1b, const float* g1, const float* b1, const float* m1, const float* v1,
    const float* c2b, const float* g2, const float* b2, const float* m2, const float* v2,
    const float* c3b, const float* g3, const float* b3, const float* m3, const float* v3,
    float* s1, float* sh1, float* s2, float* sh2, float* s3, float* sh3)
{
    int i = threadIdx.x;
    if (i < 128)      { float s = g1[i] * rsqrtf(v1[i] + 1e-5f); s1[i] = s; sh1[i] = (c1b[i] - m1[i]) * s + b1[i]; }
    else if (i < 192) { int j = i - 128; float s = g2[j] * rsqrtf(v2[j] + 1e-5f); s2[j] = s; sh2[j] = (c2b[j] - m2[j]) * s + b2[j]; }
    else              { int j = i - 192; float s = g3[j] * rsqrtf(v3[j] + 1e-5f); s3[j] = s; sh3[j] = (c3b[j] - m3[j]) * s + b3[j]; }
}

__global__ void cvt_f32_f16(const float* __restrict__ src, _Float16* __restrict__ dst, int n) {
    int i = blockIdx.x * blockDim.x + threadIdx.x;
    if (i < n) dst[i] = (_Float16)src[i];
}

// ---------------------------------------------------------------------------
// conv1 (3->128, k3) + BN + maxpool(3,2) + ReLU, fused. K=9 too skinny for WMMA.
// h1: [B][128][1024 pitch] f16, 1022 valid pooled positions.
// ---------------------------------------------------------------------------
__global__ void __launch_bounds__(256)
conv1_fused(const float* __restrict__ x, const float* __restrict__ w1,
            const float* __restrict__ scale, const float* __restrict__ shift,
            _Float16* __restrict__ h1)
{
    const int q = blockIdx.x * 256 + threadIdx.x;
    if (q >= 1022) return;
    const int oc = blockIdx.y;
    const int b  = blockIdx.z;
    const float* xb = x + (size_t)b * 3 * 2048;
    const float* w  = w1 + oc * 9;
    const float sc = scale[oc], sh = shift[oc];
    const int p0 = 2 * q;               // conv positions p0..p0+2, taps to p0+4 <= 2046
    float xv[3][5];
#pragma unroll
    for (int c = 0; c < 3; ++c)
#pragma unroll
        for (int k = 0; k < 5; ++k) xv[c][k] = xb[c * 2048 + p0 + k];
    float m = -__builtin_inff();
#pragma unroll
    for (int j = 0; j < 3; ++j) {
        float s = 0.f;
#pragma unroll
        for (int c = 0; c < 3; ++c)
#pragma unroll
            for (int k = 0; k < 3; ++k) s = fmaf(xv[c][j + k], w[c * 3 + k], s);
        m = fmaxf(m, s * sc + sh);
    }
    h1[((size_t)b * 128 + oc) * 1024 + q] = (_Float16)fmaxf(m, 0.f);
}

// ---------------------------------------------------------------------------
// Implicit-GEMM 1D conv via WMMA: Y[b][oc][p] = sum_{ic,k} W[oc][ic*KW+k]*X[b][ic][p+k]
// Cout=64 fixed (4 M-tiles). Workgroup = 8 waves -> 64 output positions (4 N-tiles,
// each wave does 2). Input tile staged in LDS; BN scale/shift in epilogue.
// ---------------------------------------------------------------------------
template<int KW>
__global__ void __launch_bounds__(256)
convgemm_wmma(const _Float16* __restrict__ W, const _Float16* __restrict__ X,
              _Float16* __restrict__ Y,
              const float* __restrict__ scale, const float* __restrict__ shift,
              int Cin, int Lin, int ldx, size_t xbs, int Lc, int ldy, size_t ybs)
{
    constexpr int WT = 72;                       // LDS row pitch >= 64 + KW - 1
    __shared__ _Float16 sx[128 * WT];            // 18.4 KB of the 320 KB/WGP
    const int b   = blockIdx.y;
    const int c0  = blockIdx.x * 64;
    const int tid = threadIdx.x;
    const int lane = tid & 31, wave = tid >> 5;
    const int ln = lane & 15, hi = lane >> 4;
    const int KK = Cin * KW;
    const int need = 64 + KW - 1;
    const _Float16* Xb = X + (size_t)b * xbs;

    for (int t = tid; t < Cin * WT; t += 256) {
        int r = t / WT, c = t - r * WT;
        int pos = c0 + c;
        _Float16 v = (_Float16)0.f;
        if (c < need && pos < Lin) v = Xb[(size_t)r * ldx + pos];
        sx[t] = v;
    }
    __syncthreads();

    const int m0 = (wave & 3) * 16;
    const _Float16* Arow = W + (size_t)(m0 + ln) * KK;
    const int nsBase = wave >> 2;
#pragma unroll
    for (int nsi = 0; nsi < 2; ++nsi) {
        const int ns = nsBase + nsi * 2;         // 0..3
        const int nloc = ns * 16 + ln;
        v8f acc = {};
        for (int k0 = 0; k0 < KK; k0 += 32) {
            __builtin_prefetch(Arow + k0 + 64, 0, 1);   // global_prefetch next A slice
            v16h a = load_a16(Arow, k0, hi);
            v16h bf;                                     // im2col gather from LDS
#pragma unroll
            for (int i = 0; i < 16; ++i) {
                int kk = k0 + hi * 16 + i;
                int ic = kk / KW;
                int kw = kk - ic * KW;
                bf[i] = sx[ic * WT + nloc + kw];
            }
            acc = __builtin_amdgcn_wmma_f32_16x16x32_f16(false, a, false, bf,
                                                         (short)0, acc, false, false);
        }
        const int pos = c0 + ns * 16 + ln;
        if (pos < Lc) {
#pragma unroll
            for (int vv = 0; vv < 8; ++vv) {
                int oc = m0 + vv + hi * 8;
                Y[(size_t)b * ybs + (size_t)oc * ldy + pos] =
                    (_Float16)(acc[vv] * scale[oc] + shift[oc]);
            }
        }
    }
}

// maxpool(k=3, stride s) + ReLU on BN-applied conv output (f16 -> f16)
__global__ void pool_relu(const _Float16* __restrict__ in, _Float16* __restrict__ out,
                          int C, int ldi, size_t ibs, int Lout, int ldo, size_t obs,
                          int s, int total)
{
    int idx = blockIdx.x * blockDim.x + threadIdx.x;
    if (idx >= total) return;
    int q = idx % Lout;
    int t = idx / Lout;
    int c = t % C;
    int b = t / C;
    const _Float16* p = in + (size_t)b * ibs + (size_t)c * ldi + (size_t)(s * q);
    float m = fmaxf(fmaxf((float)p[0], (float)p[1]), (float)p[2]);
    out[(size_t)b * obs + (size_t)c * ldo + q] = (_Float16)fmaxf(m, 0.f);
}

// ---------------------------------------------------------------------------
// fc1: [48 x 10752] @ [10752 x 256] via WMMA. 3 waves/block = 3 M-tiles; grid.x
// = 16 batch tiles. B rows are contiguous f16 activation rows -> b128 loads.
// ---------------------------------------------------------------------------
__global__ void __launch_bounds__(96)
fc1_wmma(const _Float16* __restrict__ Wh, const _Float16* __restrict__ Xh,
         const float* __restrict__ bias, float* __restrict__ Yo)
{
    const int tid = threadIdx.x;
    const int lane = tid & 31, wave = tid >> 5;
    const int ln = lane & 15, hi = lane >> 4;
    const int m0 = wave * 16;
    const int n0 = blockIdx.x * 16;
    const _Float16* Arow = Wh + (size_t)(m0 + ln) * 10752;
    const _Float16* Brow = Xh + (size_t)(n0 + ln) * 10752;
    v8f acc = {};
    for (int k0 = 0; k0 < 10752; k0 += 32) {
        v16h a = load_a16(Arow, k0, hi);
        v16h bf = load_b16_contig(Brow, k0, hi);
        acc = __builtin_amdgcn_wmma_f32_16x16x32_f16(false, a, false, bf,
                                                     (short)0, acc, false, false);
    }
#pragma unroll
    for (int vv = 0; vv < 8; ++vv) {
        int oc = m0 + vv + hi * 8;
        float v = acc[vv] + bias[oc];
        Yo[(size_t)(n0 + ln) * 48 + oc] = v > 0.f ? v : 0.f;
    }
}

// fc2 -> fc3 -> fc4 -> tanh -> theta @ basis^T. One thread per batch row.
__global__ void __launch_bounds__(256)
head_theta(const float* __restrict__ xs1,
           const float* __restrict__ w2, const float* __restrict__ b2,
           const float* __restrict__ w3, const float* __restrict__ b3,
           const float* __restrict__ w4, const float* __restrict__ b4,
           const float* __restrict__ basis, float* __restrict__ Aab)
{
    const int b = threadIdx.x;
    const float* x1 = xs1 + b * 48;
    float x2[32];
    for (int o = 0; o < 32; ++o) {
        float s = b2[o];
        for (int i = 0; i < 48; ++i) s = fmaf(w2[o * 48 + i], x1[i], s);
        x2[o] = fmaxf(s, 0.f);
    }
    float x3[16];
    for (int o = 0; o < 16; ++o) {
        float s = b3[o];
        for (int i = 0; i < 32; ++i) s = fmaf(w3[o * 32 + i], x2[i], s);
        x3[o] = fmaxf(s, 0.f);
    }
    float th[5];
    for (int o = 0; o < 5; ++o) {
        float s = b4[o];
        for (int i = 0; i < 16; ++i) s = fmaf(w4[o * 16 + i], x3[i], s);
        th[o] = tanhf(s);
    }
    for (int j = 0; j < 12; ++j) {
        float s = 0.f;
        for (int d = 0; d < 5; ++d) s = fmaf(basis[j * 5 + d], th[d], s);
        Aab[b * 12 + j] = s;          // NUM_SCALING = 0 -> no /2^N
    }
}

// ---------------------------------------------------------------------------
// CPAB exact flow integration (NC+1 = 7 iterations) + linear resample of input.
// ---------------------------------------------------------------------------
__global__ void __launch_bounds__(256)
cpab_warp(const float* __restrict__ xin, const float* __restrict__ Aab,
          float* __restrict__ out)
{
    const int i = blockIdx.x * 256 + threadIdx.x;   // 0..2047
    const int b = blockIdx.y;
    const float* Ab = Aab + b * 12;
    float aA[6], bA[6];
#pragma unroll
    for (int c = 0; c < 6; ++c) { aA[c] = Ab[2 * c]; bA[c] = Ab[2 * c + 1]; }
    float x = (float)i / 2047.0f;
    float t = 1.0f;
#pragma unroll 1
    for (int it = 0; it < 7; ++it) {
        int ci = (int)floorf(x * 6.0f);
        ci = ci < 0 ? 0 : (ci > 5 ? 5 : ci);
        float a = aA[ci], bb = bA[ci];
        float v = a * x + bb;
        float xb = (v >= 0.f) ? (float)(ci + 1) * (1.0f / 6.0f) : (float)ci * (1.0f / 6.0f);
        bool big_a = fabsf(a) > 1e-8f;
        float a_s = big_a ? a : 1.0f;
        float z  = x + bb / a_s;
        float zb = xb + bb / a_s;
        float zden = (fabsf(z) > 1e-12f) ? z : 1e-12f;
        float ratio = zb / zden;
        float t_exp = logf(fmaxf(ratio, 1e-12f)) / a_s;
        float v_s = (fabsf(v) > 1e-12f) ? v : 1.0f;
        float t_lin = (xb - x) / v_s;
        float thit = big_a ? t_exp : t_lin;
        bool valid = (fabsf(v) > 1e-12f) && (thit > 0.f) && ((big_a ? ratio : 1.0f) > 0.f);
        thit = valid ? thit : __builtin_inff();
        float tau = fminf(t, thit);
        float x_new = big_a ? (z * expf(a * tau) - (z - x)) : (x + bb * tau);
        bool hit = thit <= t;
        float nudge = (v >= 0.f) ? 1e-6f : -1e-6f;
        x = hit ? (xb + nudge) : x_new;
        x = fminf(fmaxf(x, 0.f), 1.f);
        t = fmaxf(t - tau, 0.f);
    }
    float p = fminf(fmaxf(x, 0.f), 1.f) * 2047.0f;
    int x0 = (int)floorf(p);
    x0 = x0 < 0 ? 0 : (x0 > 2046 ? 2046 : x0);
    float w = p - (float)x0;
    const float* xb0 = xin + (size_t)b * 3 * 2048;
#pragma unroll
    for (int ch = 0; ch < 3; ++ch) {
        float d0 = xb0[ch * 2048 + x0];
        float d1 = xb0[ch * 2048 + x0 + 1];
        out[((size_t)b * 3 + ch) * 2048 + i] = d0 * (1.f - w) + d1 * w;
    }
}

// ---------------------------------------------------------------------------
extern "C" void kernel_launch(void* const* d_in, const int* in_sizes, int n_in,
                              void* d_out, int out_size, void* d_ws, size_t ws_size,
                              hipStream_t stream)
{
    (void)in_sizes; (void)n_in; (void)out_size; (void)ws_size;
    const float* x       = (const float*)d_in[0];
    const float* conv1_w = (const float*)d_in[1];
    const float* conv1_b = (const float*)d_in[2];
    const float* bn1_g   = (const float*)d_in[3];
    const float* bn1_b   = (const float*)d_in[4];
    const float* bn1_m   = (const float*)d_in[5];
    const float* bn1_v   = (const float*)d_in[6];
    const float* conv2_w = (const float*)d_in[7];
    const float* conv2_b = (const float*)d_in[8];
    const float* bn2_g   = (const float*)d_in[9];
    const float* bn2_b   = (const float*)d_in[10];
    const float* bn2_m   = (const float*)d_in[11];
    const float* bn2_v   = (const float*)d_in[12];
    const float* conv3_w = (const float*)d_in[13];
    const float* conv3_b = (const float*)d_in[14];
    const float* bn3_g   = (const float*)d_in[15];
    const float* bn3_b   = (const float*)d_in[16];
    const float* bn3_m   = (const float*)d_in[17];
    const float* bn3_v   = (const float*)d_in[18];
    const float* fc1_w   = (const float*)d_in[19];
    const float* fc1_b   = (const float*)d_in[20];
    const float* fc2_w   = (const float*)d_in[21];
    const float* fc2_b   = (const float*)d_in[22];
    const float* fc3_w   = (const float*)d_in[23];
    const float* fc3_b   = (const float*)d_in[24];
    const float* fc4_w   = (const float*)d_in[25];
    const float* fc4_b   = (const float*)d_in[26];
    const float* basis   = (const float*)d_in[27];

    char* ws = (char*)d_ws;
    size_t off = 0;
    auto take = [&](size_t bytes) -> void* {
        void* p = ws + off;
        off += (bytes + 255) & ~(size_t)255;
        return p;
    };
    float* s1  = (float*)take(128 * 4);
    float* sh1 = (float*)take(128 * 4);
    float* s2  = (float*)take(64 * 4);
    float* sh2 = (float*)take(64 * 4);
    float* s3  = (float*)take(64 * 4);
    float* sh3 = (float*)take(64 * 4);
    _Float16* w2h  = (_Float16*)take((size_t)64 * 640 * 2);
    _Float16* w3h  = (_Float16*)take((size_t)64 * 192 * 2);
    _Float16* fc1h = (_Float16*)take((size_t)48 * 10752 * 2);
    float* xs1 = (float*)take((size_t)256 * 48 * 4);
    float* Aab = (float*)take((size_t)256 * 12 * 4);
    _Float16* h1 = (_Float16*)take((size_t)256 * 128 * 1024 * 2);  // 67 MB
    _Float16* c2 = (_Float16*)take((size_t)256 * 64 * 1024 * 2);   // 33.5 MB
    _Float16* h2 = (_Float16*)take((size_t)256 * 64 * 344 * 2);    // 11.3 MB
    _Float16* c3 = h1;   // reuse: h1 dead after conv2 GEMM
    _Float16* h3 = c2;   // reuse: c2 dead after pool2

    prep_scales<<<1, 256, 0, stream>>>(conv1_b, bn1_g, bn1_b, bn1_m, bn1_v,
                                       conv2_b, bn2_g, bn2_b, bn2_m, bn2_v,
                                       conv3_b, bn3_g, bn3_b, bn3_m, bn3_v,
                                       s1, sh1, s2, sh2, s3, sh3);
    cvt_f32_f16<<<(64 * 640 + 255) / 256, 256, 0, stream>>>(conv2_w, w2h, 64 * 640);
    cvt_f32_f16<<<(64 * 192 + 255) / 256, 256, 0, stream>>>(conv3_w, w3h, 64 * 192);
    cvt_f32_f16<<<(48 * 10752 + 255) / 256, 256, 0, stream>>>(fc1_w, fc1h, 48 * 10752);

    conv1_fused<<<dim3(4, 128, 256), 256, 0, stream>>>(x, conv1_w, s1, sh1, h1);

    convgemm_wmma<5><<<dim3(16, 256), 256, 0, stream>>>(
        w2h, h1, c2, s2, sh2, 128, 1022, 1024, (size_t)128 * 1024, 1018, 1024, (size_t)64 * 1024);
    pool_relu<<<(256 * 64 * 339 + 255) / 256, 256, 0, stream>>>(
        c2, h2, 64, 1024, (size_t)64 * 1024, 339, 344, (size_t)64 * 344, 3, 256 * 64 * 339);

    convgemm_wmma<3><<<dim3(6, 256), 256, 0, stream>>>(
        w3h, h2, c3, s3, sh3, 64, 339, 344, (size_t)64 * 344, 337, 352, (size_t)64 * 352);
    pool_relu<<<(256 * 64 * 168 + 255) / 256, 256, 0, stream>>>(
        c3, h3, 64, 352, (size_t)64 * 352, 168, 168, (size_t)64 * 168, 2, 256 * 64 * 168);

    fc1_wmma<<<16, 96, 0, stream>>>(fc1h, h3, fc1_b, xs1);
    head_theta<<<1, 256, 0, stream>>>(xs1, fc2_w, fc2_b, fc3_w, fc3_b, fc4_w, fc4_b, basis, Aab);
    cpab_warp<<<dim3(8, 256), 256, 0, stream>>>(x, Aab, (float*)d_out);
}
